// DeformableTransformer_54391465837191
// MI455X (gfx1250) — matrix operand order repaired and checked by reading, not compiled, not run on hardware
//
#include <hip/hip_runtime.h>
#include <hip/hip_bf16.h>

// ---------------- static problem config (matches reference) ----------------
#define LQ_TOT 13294
#define BB     2
#define CC     256
#define HH     8
#define LL     4
#define PP     4
#define DHH    32
#define DFFN   2048
#define M_TOT  (BB * LQ_TOT)   // 26588
#define M_PAD  26592           // 16 * 1662
#define M_TILES (M_PAD / 16)   // 1662

typedef __bf16 bf16_t;
typedef __attribute__((ext_vector_type(16))) __bf16 v16bf;
typedef __attribute__((ext_vector_type(8)))  float  v8f;
typedef __attribute__((ext_vector_type(4)))  unsigned int u32x4;
typedef __attribute__((ext_vector_type(8)))  int i32x8;
typedef __attribute__((ext_vector_type(4)))  int i32x4;

struct alignas(16) Frag8 { unsigned u[4]; };          // 8 bf16 = 16 bytes
union AFrag { v16bf v; Frag8 h[2]; };

#if defined(__gfx1250__) && __has_builtin(__builtin_amdgcn_tensor_load_to_lds) && \
    __has_builtin(__builtin_amdgcn_s_wait_tensorcnt)
#define HAVE_TDM 1
#else
#define HAVE_TDM 0
#endif

static __device__ __forceinline__ bf16_t f2bf(float f) {
  unsigned u = __builtin_bit_cast(unsigned, f);
  unsigned r = (u + 0x7FFFu + ((u >> 16) & 1u)) >> 16;   // RNE
  return __builtin_bit_cast(bf16_t, (unsigned short)r);
}
static __device__ __forceinline__ float bf2f(bf16_t b) {
  unsigned u = ((unsigned)__builtin_bit_cast(unsigned short, b)) << 16;
  return __builtin_bit_cast(float, u);
}

// ---------------- prep: q = src + pos, bf16 casts, zero pad rows -----------
__global__ __launch_bounds__(256) void k_prep(const float* __restrict__ src,
                                              const float* __restrict__ pos,
                                              bf16_t* __restrict__ src_bf,
                                              bf16_t* __restrict__ q_bf) {
  int tid = blockIdx.x * 256 + threadIdx.x;
  if (tid >= M_PAD * CC) return;
  int m = tid >> 8;
  float s = 0.f, p = 0.f;
  if (m < M_TOT) { s = src[tid]; p = pos[tid]; }
  src_bf[tid] = f2bf(s);
  q_bf[tid]   = f2bf(s + p);
}

// ------- pack fp32 weight [K,N] into grouped WMMA B-fragment order ---------
// layout: [ntg = n/64][kt][j = ntile in group (4)][lane(32)][e(16)]
//   -> per (ntg,kt): contiguous 4KB panel (what one block consumes per K step)
//   lane = (khalf<<4)|ncol, K = kt*32 + khalf*16 + e, n = (ntg*4+j)*16 + ncol
__global__ __launch_bounds__(256) void k_pack_w(const float* __restrict__ w,
                                                bf16_t* __restrict__ wp,
                                                int K, int N) {
  int tid = blockIdx.x * 256 + threadIdx.x;
  if (tid >= K * N) return;
  int e    = tid & 15;
  int lane = (tid >> 4) & 31;
  int j    = (tid >> 9) & 3;
  int rest = tid >> 11;
  int Kt   = K >> 5;
  int kt   = rest % Kt;
  int ntg  = rest / Kt;
  int k = kt * 32 + ((lane >> 4) << 4) + e;
  int n = (ntg * 4 + j) * 16 + (lane & 15);
  wp[tid] = f2bf(w[(size_t)k * N + n]);
}

// ---------------- bf16 WMMA GEMM: out = A[Mpad,K] @ W[K,N] + bias ----------
// one wave -> 16x64 tile (4 n-tiles), 8 waves/block over M tiles.
// STAGE=true: whole 4KB*Ktiles B panel staged into LDS once per block
//             (Tensor Data Mover if available), inner loop reads via DS.
template <bool STAGE>
__global__ __launch_bounds__(256) void k_gemm_bf16(
    const bf16_t* __restrict__ A,      // [M_PAD, K] row-major
    const bf16_t* __restrict__ Wp,     // packed grouped B fragments
    const float*  __restrict__ bias,   // [N]
    float*  __restrict__ outF,         // [M_PAD, N] or null
    bf16_t* __restrict__ outB,         // [M_PAD, N] or null
    int N, int K, int relu) {
  const int lane = threadIdx.x & 31;
  const int wid  = threadIdx.x >> 5;
  const int mt   = blockIdx.x * 8 + wid;
  const int half = lane >> 4;
  const int mrow = lane & 15;
  const int Ktiles = K >> 5;

  const bf16_t* __restrict__ bsrc;     // B fragment source (global or LDS)

  __shared__ bf16_t smem[STAGE ? 8 * 2048 : 1];  // 32KB panel for K<=256
  if constexpr (STAGE) {
#if HAVE_TDM
    if (wid == 0) {
      // Tensor DMA descriptor (ISA ch.8): 2D tile, 4-byte elements,
      // tile_dim0 = 1024 dwords (4KB row), tile_dim1 = Ktiles rows,
      // tensor_dim0_stride = 1024 (contiguous panel), dest LDS offset 0.
      unsigned long long ga =
          (unsigned long long)(const void*)(Wp + (size_t)blockIdx.y * Ktiles * 2048);
      u32x4 g0;
      g0[0] = 1u;                                            // count = 1
      g0[1] = 0u;                                            // lds_addr = 0
      g0[2] = (unsigned)(ga & 0xFFFFFFFFull);                // global_addr lo
      g0[3] = (unsigned)((ga >> 32) & 0x1FFFFFFull) | (2u << 30);  // hi | type=2
      i32x8 g1;
      g1[0] = (int)(2u << 16);                   // data_size = 2 (4B), mask=0
      g1[1] = (int)(1024u << 16);                // tensor_dim0 lo16 = 1024
      g1[2] = (int)((unsigned)Ktiles << 16);     // dim0 hi=0 | tensor_dim1 lo16
      g1[3] = (int)(1024u << 16);                // dim1 hi=0 | tile_dim0 = 1024
      g1[4] = Ktiles;                            // tile_dim1 | tile_dim2=0
      g1[5] = 1024;                              // tensor_dim0_stride lo32
      g1[6] = 0;
      g1[7] = 0;
      i32x4 gz4 = {0, 0, 0, 0};
      i32x8 gz8 = {0, 0, 0, 0, 0, 0, 0, 0};
      __builtin_amdgcn_tensor_load_to_lds(g0, g1, gz4, gz4, gz8, 0);
      __builtin_amdgcn_s_wait_tensorcnt(0);
    }
#else
    {  // cooperative fallback copy: 256 threads x 16B chunks
      const bf16_t* src = Wp + (size_t)blockIdx.y * Ktiles * 2048;
      for (int i = threadIdx.x; i < Ktiles * 128; i += 256)
        *(Frag8*)(smem + i * 16) = *(const Frag8*)(src + i * 16);
    }
#endif
    __syncthreads();
    bsrc = smem + lane * 16;
  } else {
    bsrc = Wp + (size_t)blockIdx.y * Ktiles * 2048 + lane * 16;
  }

  if (mt >= M_TILES) return;

  const bf16_t* aptr = A + (size_t)(mt * 16 + mrow) * K + half * 16;
  v8f acc[4] = {};

  for (int kt = 0; kt < Ktiles; ++kt) {
    AFrag a, b0, b1, b2, b3;
    a.h[0] = *(const Frag8*)(aptr);
    a.h[1] = *(const Frag8*)(aptr + 8);
    if (kt + 1 < Ktiles)
      __builtin_prefetch(aptr + 32, 0, 1);         // global_prefetch_b8
    // issue all B-fragment loads up front (immediate offsets off one base)
    b0.h[0] = *(const Frag8*)(bsrc + 0);
    b0.h[1] = *(const Frag8*)(bsrc + 8);
    b1.h[0] = *(const Frag8*)(bsrc + 512);
    b1.h[1] = *(const Frag8*)(bsrc + 520);
    b2.h[0] = *(const Frag8*)(bsrc + 1024);
    b2.h[1] = *(const Frag8*)(bsrc + 1032);
    b3.h[0] = *(const Frag8*)(bsrc + 1536);
    b3.h[1] = *(const Frag8*)(bsrc + 1544);
    acc[0] = __builtin_amdgcn_wmma_f32_16x16x32_bf16(false, a.v, false, b0.v,
                                                     (short)0, acc[0], false, false);
    acc[1] = __builtin_amdgcn_wmma_f32_16x16x32_bf16(false, a.v, false, b1.v,
                                                     (short)0, acc[1], false, false);
    acc[2] = __builtin_amdgcn_wmma_f32_16x16x32_bf16(false, a.v, false, b2.v,
                                                     (short)0, acc[2], false, false);
    acc[3] = __builtin_amdgcn_wmma_f32_16x16x32_bf16(false, a.v, false, b3.v,
                                                     (short)0, acc[3], false, false);
    aptr += 32;
    bsrc += 2048;
  }

  const int mbase = mt * 16 + (half ? 8 : 0);
#pragma unroll
  for (int j = 0; j < 4; ++j) {
    const int n0 = blockIdx.y * 64 + j * 16 + mrow;
    const float bv = bias ? bias[n0] : 0.f;
#pragma unroll
    for (int r = 0; r < 8; ++r) {
      const int m = mbase + r;
      if (m < M_TOT) {
        float v = acc[j][r] + bv;
        if (relu) v = fmaxf(v, 0.f);
        if (outF) outF[(size_t)m * N + n0] = v;
        if (outB) outB[(size_t)m * N + n0] = f2bf(v);
      } else if (outB) {                       // zero pad rows for next GEMM
        outB[(size_t)m * N + n0] = f2bf(0.f);
      }
    }
  }
}

// ---------------- softmax over the 16 (L*P) logits per (token, head) ------
__global__ __launch_bounds__(256) void k_softmax(float* __restrict__ attn) {
  int tid = blockIdx.x * 256 + threadIdx.x;
  if (tid >= M_TOT * HH) return;
  float* a = attn + (size_t)tid * 16;
  float mx = -1e30f;
#pragma unroll
  for (int i = 0; i < 16; ++i) mx = fmaxf(mx, a[i]);
  float e[16], s = 0.f;
#pragma unroll
  for (int i = 0; i < 16; ++i) { e[i] = expf(a[i] - mx); s += e[i]; }
  float inv = 1.f / s;
#pragma unroll
  for (int i = 0; i < 16; ++i) a[i] = e[i] * inv;
}

// ---------------- multi-scale deformable sampling --------------------------
// block = one token m; wave = head h; lane = dh channel (coalesced gathers)
__global__ __launch_bounds__(256) void k_sample(
    const bf16_t* __restrict__ value,  // [B*Lv, 256] bf16
    const float*  __restrict__ off,    // [M, 256] fp32
    const float*  __restrict__ attn,   // [M, 128] fp32 (softmaxed)
    const float*  __restrict__ refp,   // [M, L, 2]
    bf16_t* __restrict__ out) {        // [M_PAD, 256] bf16
  const int lane = threadIdx.x & 31;
  const int h    = threadIdx.x >> 5;
  const int m    = blockIdx.x;
  if (m >= M_TOT) { out[(size_t)m * CC + h * DHH + lane] = f2bf(0.f); return; }

  const int levW[4] = {100, 50, 25, 13};
  const int levH[4] = {100, 50, 25, 13};
  const int levS[4] = {0, 10000, 12500, 13125};

  const int b = m / LQ_TOT;
  const float* om = off  + (size_t)m * CC + h * 32;
  const float* am = attn + (size_t)m * 128 + h * 16;
  const float* rm = refp + (size_t)m * 8;
  const bf16_t* vb = value + ((size_t)b * LQ_TOT) * CC + h * DHH + lane;

  float acc = 0.f;
#pragma unroll
  for (int l = 0; l < 4; ++l) {
    const int Wi = levW[l], Hi = levH[l];
    const float rx = rm[l * 2 + 0], ry = rm[l * 2 + 1];
#pragma unroll
    for (int p = 0; p < 4; ++p) {
      const int pi = l * 4 + p;
      const float gx = rx * (float)Wi + om[pi * 2 + 0] - 0.5f;
      const float gy = ry * (float)Hi + om[pi * 2 + 1] - 0.5f;
      const float aw = am[pi];
      const float x0f = floorf(gx), y0f = floorf(gy);
      const int x0 = (int)x0f, y0 = (int)y0f;
      const float fx = gx - x0f, fy = gy - y0f;
#pragma unroll
      for (int dy = 0; dy < 2; ++dy) {
#pragma unroll
        for (int dx = 0; dx < 2; ++dx) {
          const int xi = x0 + dx, yi = y0 + dy;
          const float w = (dx ? fx : 1.f - fx) * (dy ? fy : 1.f - fy) * aw;
          if (xi >= 0 && xi < Wi && yi >= 0 && yi < Hi) {
            acc += w * bf2f(vb[(size_t)(levS[l] + yi * Wi + xi) * CC]);
          }
        }
      }
    }
  }
  out[(size_t)m * CC + h * DHH + lane] = f2bf(acc);
}

// ---------------- x = LayerNorm(a + res); wave32 shuffle reduction ---------
__global__ __launch_bounds__(256) void k_ln(
    const float* __restrict__ a, const float* __restrict__ res,
    const float* __restrict__ g, const float* __restrict__ bb,
    float* __restrict__ xf, bf16_t* __restrict__ xb) {
  const int lane = threadIdx.x & 31;
  const int wid  = threadIdx.x >> 5;
  const int row  = blockIdx.x * 8 + wid;
  if (row >= M_PAD) return;
  if (row >= M_TOT) {                        // zero bf16 pad rows (GEMM input)
    if (xb)
#pragma unroll
      for (int i = 0; i < 8; ++i) xb[(size_t)row * CC + i * 32 + lane] = f2bf(0.f);
    return;
  }
  float v[8], s = 0.f;
#pragma unroll
  for (int i = 0; i < 8; ++i) {
    const int c = i * 32 + lane;
    v[i] = a[(size_t)row * CC + c] + res[(size_t)row * CC + c];
    s += v[i];
  }
#pragma unroll
  for (int o = 16; o > 0; o >>= 1) s += __shfl_xor(s, o, 32);
  const float mean = s * (1.f / 256.f);
  float q = 0.f;
#pragma unroll
  for (int i = 0; i < 8; ++i) { const float d = v[i] - mean; q += d * d; }
#pragma unroll
  for (int o = 16; o > 0; o >>= 1) q += __shfl_xor(q, o, 32);
  const float inv = rsqrtf(q * (1.f / 256.f) + 1e-5f);
#pragma unroll
  for (int i = 0; i < 8; ++i) {
    const int c = i * 32 + lane;
    const float y = (v[i] - mean) * inv * g[c] + bb[c];
    if (xf) xf[(size_t)row * CC + c] = y;
    if (xb) xb[(size_t)row * CC + c] = f2bf(y);
  }
}

// ---------------------------------------------------------------------------
extern "C" void kernel_launch(void* const* d_in, const int* in_sizes, int n_in,
                              void* d_out, int out_size, void* d_ws, size_t ws_size,
                              hipStream_t stream) {
  (void)in_sizes; (void)n_in; (void)out_size; (void)ws_size;
  const float* src     = (const float*)d_in[0];
  const float* pos     = (const float*)d_in[1];
  const float* refp    = (const float*)d_in[2];
  const float* w_value = (const float*)d_in[3];
  const float* b_value = (const float*)d_in[4];
  const float* w_off   = (const float*)d_in[5];
  const float* b_off   = (const float*)d_in[6];
  const float* w_attn  = (const float*)d_in[7];
  const float* b_attn  = (const float*)d_in[8];
  const float* w_out   = (const float*)d_in[9];
  const float* b_out   = (const float*)d_in[10];
  const float* ln1_g   = (const float*)d_in[11];
  const float* ln1_b   = (const float*)d_in[12];
  const float* w_ffn1  = (const float*)d_in[13];
  const float* b_ffn1  = (const float*)d_in[14];
  const float* w_ffn2  = (const float*)d_in[15];
  const float* b_ffn2  = (const float*)d_in[16];
  const float* ln2_g   = (const float*)d_in[17];
  const float* ln2_b   = (const float*)d_in[18];

  char* ws = (char*)d_ws;
  size_t cur = 0;
  auto alloc = [&](size_t bytes) -> void* {
    void* p = ws + cur;
    cur = (cur + bytes + 255) & ~(size_t)255;
    return p;
  };
  bf16_t* src_bf   = (bf16_t*)alloc((size_t)M_PAD * CC * 2);
  bf16_t* q_bf     = (bf16_t*)alloc((size_t)M_PAD * CC * 2);
  bf16_t* value_bf = (bf16_t*)alloc((size_t)M_PAD * CC * 2);
  float*  off_f    = (float*) alloc((size_t)M_PAD * CC * 4);
  float*  attn_f   = (float*) alloc((size_t)M_PAD * 128 * 4);
  bf16_t* samp_bf  = (bf16_t*)alloc((size_t)M_PAD * CC * 2);
  float*  src2_f   = (float*) alloc((size_t)M_PAD * CC * 4);
  float*  x_f      = (float*) alloc((size_t)M_PAD * CC * 4);
  bf16_t* x_bf     = (bf16_t*)alloc((size_t)M_PAD * CC * 2);
  bf16_t* h_bf     = (bf16_t*)alloc((size_t)M_PAD * DFFN * 2);
  float*  ff_f     = (float*) alloc((size_t)M_PAD * CC * 4);
  bf16_t* wp_value = (bf16_t*)alloc((size_t)CC * CC * 2);
  bf16_t* wp_off   = (bf16_t*)alloc((size_t)CC * CC * 2);
  bf16_t* wp_attn  = (bf16_t*)alloc((size_t)CC * 128 * 2);
  bf16_t* wp_out   = (bf16_t*)alloc((size_t)CC * CC * 2);
  bf16_t* wp_ffn1  = (bf16_t*)alloc((size_t)CC * DFFN * 2);
  bf16_t* wp_ffn2  = (bf16_t*)alloc((size_t)DFFN * CC * 2);

  // 1) activation prep
  k_prep<<<dim3(M_PAD), dim3(256), 0, stream>>>(src, pos, src_bf, q_bf);

  // 2) weight packing (tiny)
  auto pack = [&](const float* w, bf16_t* wp, int K, int N) {
    k_pack_w<<<dim3((K * N + 255) / 256), dim3(256), 0, stream>>>(w, wp, K, N);
  };
  pack(w_value, wp_value, CC, CC);
  pack(w_off,   wp_off,   CC, CC);
  pack(w_attn,  wp_attn,  CC, 128);
  pack(w_out,   wp_out,   CC, CC);
  pack(w_ffn1,  wp_ffn1,  CC, DFFN);
  pack(w_ffn2,  wp_ffn2,  DFFN, CC);

  const dim3 gblk(256);
  const unsigned gx = (M_TILES + 7) / 8;
  // K=256 GEMMs: B panel (32KB) staged via TDM into LDS
  auto gemmS = [&](const bf16_t* A, const bf16_t* Wp, const float* bias,
                   float* oF, bf16_t* oB, int N, int relu) {
    k_gemm_bf16<true><<<dim3(gx, N / 64), gblk, 0, stream>>>(A, Wp, bias, oF, oB,
                                                             N, CC, relu);
  };

  // 3) projections
  gemmS(src_bf, wp_value, b_value, nullptr, value_bf, CC, 0);
  gemmS(q_bf,   wp_off,   b_off,   off_f,   nullptr,  CC, 0);
  gemmS(q_bf,   wp_attn,  b_attn,  attn_f,  nullptr,  128, 0);

  // 4) softmax over L*P per (token, head)
  k_softmax<<<dim3((M_TOT * HH + 255) / 256), dim3(256), 0, stream>>>(attn_f);

  // 5) deformable bilinear sampling
  k_sample<<<dim3(M_PAD), dim3(256), 0, stream>>>(value_bf, off_f, attn_f, refp, samp_bf);

  // 6) output projection + residual LN1
  gemmS(samp_bf, wp_out, b_out, src2_f, nullptr, CC, 0);
  k_ln<<<dim3((M_PAD + 7) / 8), dim3(256), 0, stream>>>(src2_f, src, ln1_g, ln1_b, x_f, x_bf);

  // 7) FFN (ffn1 staged K=256; ffn2 K=2048 uses direct-global B path)
  gemmS(x_bf, wp_ffn1, b_ffn1, nullptr, h_bf, DFFN, 1);      // relu -> bf16
  k_gemm_bf16<false><<<dim3(gx, CC / 64), gblk, 0, stream>>>(h_bf, wp_ffn2, b_ffn2,
                                                             ff_f, nullptr, CC, DFFN, 0);

  // 8) final residual LN2 -> d_out (fp32)
  k_ln<<<dim3((M_PAD + 7) / 8), dim3(256), 0, stream>>>(ff_f, x_f, ln2_g, ln2_b,
                                                        (float*)d_out, nullptr);
}
